// Decoder_13151189860783
// MI455X (gfx1250) — compile-verified, matrix operand order checked by reading, not run on hardware
//
#include <hip/hip_runtime.h>
#include <hip/hip_bf16.h>
#include <math.h>

#define B_ 512
#define N_ 200
#define D_ 128
#define H_ 8
#define DH_ 16
#define CLIPV 10.0f
#define NEGV (-1e9f)
#define SPAD 136   // padded LDS row stride (halves) for K1/V/K2

typedef __attribute__((ext_vector_type(16))) _Float16 v16h;
typedef __attribute__((ext_vector_type(8)))  float    v8f;

union frag16 { v16h v; int4 q[2]; };

// ---------------- prep kernels ----------------

__global__ void k_cvt_ne(const float* __restrict__ ne, _Float16* __restrict__ neh, int n) {
  int i = blockIdx.x * blockDim.x + threadIdx.x;
  int stride = gridDim.x * blockDim.x;
  for (; i < n; i += stride) neh[i] = (_Float16)ne[i];
}

// Pre-permute weights into exact WMMA B-fragment order:
//   wt[w][kb][ct][lane][e] = (f16) W_w[col][k0+p]
// with kb0=(lane>>4)*8, col=ct*16+(lane&15), e=2*ii+p,
//      k0 = kb*32 + (ii<4 ? kb0+2*ii : kb0+16+2*(ii-4)).
// Then each B fragment is two aligned 16B LDS loads, no packing moves.
__global__ void k_transpose_w(const float* __restrict__ wk1, const float* __restrict__ wv,
                              const float* __restrict__ wk2, _Float16* __restrict__ wt) {
  int i = blockIdx.x * blockDim.x + threadIdx.x;
  if (i >= 3 * 16384) return;
  int w = i >> 14; int r = i & 16383;
  int e    = r & 15;
  int lane = (r >> 4) & 31;
  int ct   = (r >> 9) & 7;
  int kb   = (r >> 12) & 3;
  int kb0  = (lane >> 4) * 8;
  int ii   = e >> 1;
  int p    = e & 1;
  int k0   = kb * 32 + ((ii < 4) ? (kb0 + 2 * ii) : (kb0 + 16 + 2 * (ii - 4))) + p;
  int col  = ct * 16 + (lane & 15);
  const float* src = (w == 0) ? wk1 : ((w == 1) ? wv : wk2);
  wt[i] = (_Float16)src[col * 128 + k0];
}

// ---------------- static projection GEMMs (WMMA) ----------------
// K1 = NE @ Wk1.T + b ; V = NE @ Wv.T ; K2 = NE @ Wk2.T  (f16 out, f32 accumulate)
// One wave computes a 16x128 output strip: 4 A fragments reused across 8 column
// tiles -> 32 v_wmma per wave. Weights async-staged into LDS (32KB) per block.
__global__ __launch_bounds__(128) void k_gemm_static(
    const _Float16* __restrict__ neh,   // [B*N, 128]
    const _Float16* __restrict__ wt,    // 3 x 16384 halves, fragment-ordered
    const float*    __restrict__ bias,  // Wk1_b [128]
    _Float16*       __restrict__ kh)    // 3 x [B*N, 128]
{
  __shared__ _Float16 sW[128 * 128];

  const int wm = blockIdx.z;                  // which matrix 0..2
  // async stage weight matrix into LDS (gfx1250 ASYNCcnt path)
  {
    const unsigned long long gw =
        (unsigned long long)(uintptr_t)wt + (unsigned long long)wm * 32768ull;
    for (int i = threadIdx.x; i < 2048; i += 128) {
      unsigned off = (unsigned)(i * 16);
      unsigned dst = (unsigned)(uintptr_t)(sW + i * 8);
      asm volatile("global_load_async_to_lds_b128 %0, %1, %2"
                   :: "v"(dst), "v"(off), "s"(gw) : "memory");
    }
  }

  const int lane = threadIdx.x & 31;
  const int wid  = threadIdx.x >> 5;
  const int rt   = blockIdx.x * 4 + wid;      // row tile 0..6399
  const int mrow = lane & 15;
  const int hi   = lane >> 4;                 // 0 or 1
  const int kb0  = hi * 8;

  const _Float16* Arow = neh + (size_t)(rt * 16 + mrow) * 128;

  // A fragments (16-bit A 16x32 layout): two contiguous 8-half runs per lane
  frag16 afrag[4];
  #pragma unroll
  for (int kb = 0; kb < 4; ++kb) {
    afrag[kb].q[0] = *(const int4*)(Arow + kb * 32 + kb0);
    afrag[kb].q[1] = *(const int4*)(Arow + kb * 32 + kb0 + 16);
  }

  asm volatile("s_wait_asynccnt 0" ::: "memory");   // weights resident in LDS
  __syncthreads();

  _Float16* outb = kh + (size_t)wm * ((size_t)B_ * N_ * D_);

  #pragma unroll
  for (int ct = 0; ct < 8; ++ct) {
    const int col = ct * 16 + mrow;
    v8f c = {};
    #pragma unroll
    for (int kb = 0; kb < 4; ++kb) {
      const _Float16* wf = sW + ((((kb << 3) | ct) << 5) + lane) * 16;
      frag16 bu;
      bu.q[0] = *(const int4*)(wf);
      bu.q[1] = *(const int4*)(wf + 8);
      c = __builtin_amdgcn_wmma_f32_16x16x32_f16(false, afrag[kb].v, false, bu.v,
                                                 (short)0, c, false, false);
    }
    const float bv = (wm == 0) ? bias[col] : 0.f;   // hoisted: one load per strip
    #pragma unroll
    for (int i = 0; i < 8; ++i) {
      int m = i + hi * 8;   // C/D layout: vgpr i holds M = i (+8 for upper half-wave)
      outb[(size_t)(rt * 16 + m) * 128 + col] = (_Float16)(c[i] + bv);
    }
  }
}

// ---------------- per-batch sequential decode ----------------
// One workgroup (256 thr) per batch element; K1/V/K2 (f16) staged into 320KB WGP
// LDS via gfx1250 async-to-LDS loads (ASYNCcnt path).

extern __shared__ char smem[];

__global__ __launch_bounds__(256) void k_decode(
    const _Float16* __restrict__ kh,     // 3 x [B*N,128] f16: K1, V, K2
    const float* __restrict__ ne,        // node_embeddings f32 [B,N,128]
    const float* __restrict__ rf,        // [B,2]
    const float* __restrict__ wq,        // [128,130]
    const float* __restrict__ wout,      // [128,128]
    const float* __restrict__ initnode,  // [128]
    const float* __restrict__ x,         // [2,B,200,200]
    float* __restrict__ out)             // cost[512] f1[512] f2[512] ll[512] pi[512*200]
{
  const int b = blockIdx.x;
  const int tid = threadIdx.x;

  _Float16* sK1    = (_Float16*)(smem);            // 200*136*2 = 54400
  _Float16* sV     = (_Float16*)(smem + 54400);
  _Float16* sK2    = (_Float16*)(smem + 108800);
  float*    sAttn  = (float*)(smem + 163200);      // [8][200]
  float*    sLogits= (float*)(smem + 169600);      // [200]
  float*    sQ     = (float*)(smem + 170400);      // [128]
  float*    sMha   = (float*)(smem + 170912);      // [128]
  float*    sQ2    = (float*)(smem + 171424);      // [128]
  float*    sCtx   = (float*)(smem + 171936);      // [132]
  int*      sMask  = (int*)(smem + 172464);        // [200]
  int*      sPi    = (int*)(smem + 173264);        // [200]
  float*    sChosen= (float*)(smem + 174064);      // [200]
  int*      sSol   = (int*)(smem + 174864);        // [200]
  float*    sRedF  = (float*)(smem + 175664);      // [256]
  int*      sRedI  = (int*)(smem + 176688);        // [256]
  int*      sScal  = (int*)(smem + 177712);        // scalars

  // --- async stage K1/V/K2 into LDS (padded rows), gfx1250 ASYNCcnt path ---
  const size_t kbase = (size_t)b * (N_ * D_);
  const unsigned long long g1 = (unsigned long long)(uintptr_t)kh;       // K1 base
  const unsigned long long g2 = g1 + 26214400ull;                        // V base
  const unsigned long long g3 = g1 + 52428800ull;                        // K2 base
  for (int c = tid; c < N_ * 16; c += 256) {
    int n = c >> 4, j = c & 15;
    unsigned off = (unsigned)((kbase + (size_t)n * 128 + j * 8) * 2);    // byte offset
    unsigned d1 = (unsigned)(uintptr_t)(sK1 + n * SPAD + j * 8);
    unsigned d2 = (unsigned)(uintptr_t)(sV  + n * SPAD + j * 8);
    unsigned d3 = (unsigned)(uintptr_t)(sK2 + n * SPAD + j * 8);
    asm volatile("global_load_async_to_lds_b128 %0, %1, %2"
                 :: "v"(d1), "v"(off), "s"(g1) : "memory");
    asm volatile("global_load_async_to_lds_b128 %0, %1, %2"
                 :: "v"(d2), "v"(off), "s"(g2) : "memory");
    asm volatile("global_load_async_to_lds_b128 %0, %1, %2"
                 :: "v"(d3), "v"(off), "s"(g3) : "memory");
  }
  for (int n = tid; n < N_; n += 256) sMask[n] = 0;
  if (tid < D_) sCtx[2 + tid] = initnode[tid];
  if (tid == 0) { sCtx[0] = rf[b * 2 + 0]; sCtx[1] = rf[b * 2 + 1]; sScal[0] = 1; }

  const float* x0b = x + (size_t)b * 40000;
  const float* x1b = x + (size_t)(B_ + b) * 40000;
  // warm L2 with the cost matrices while the 200-step loop runs (global_prefetch_b8)
  for (int n = tid; n < N_; n += 256) {
    __builtin_prefetch(x0b + n * 200, 0, 1);
    __builtin_prefetch(x1b + n * 200, 0, 1);
  }
  asm volatile("s_wait_asynccnt 0" ::: "memory");   // drain async-to-LDS copies
  __syncthreads();

  for (int t = 0; t < N_; ++t) {
    // Q1[j] = sum_i Wq[j,i] * ctx[i]
    if (tid < D_) {
      const float* wr = wq + tid * 130;
      float acc = 0.f;
      for (int i = 0; i < 130; ++i) acc += wr[i] * sCtx[i];
      sQ[tid] = acc;
    }
    __syncthreads();
    // scores[h][n] = q_h . K1h[n] / 4, masked
    for (int p = tid; p < H_ * N_; p += 256) {
      int h = p / N_, n = p - h * N_;
      const _Float16* kr = sK1 + n * SPAD + h * DH_;
      float s = 0.f;
      #pragma unroll
      for (int d = 0; d < DH_; ++d) s += (float)kr[d] * sQ[h * DH_ + d];
      s *= 0.25f;
      if (sMask[n]) s = NEGV;
      sAttn[h * N_ + n] = s;
    }
    __syncthreads();
    // softmax per head: wave w handles head w (wave32 shuffles)
    {
      int wvi = tid >> 5, lane = tid & 31;
      if (wvi < H_) {
        float* row = sAttn + wvi * N_;
        float mx = -1e30f;
        for (int n = lane; n < N_; n += 32) mx = fmaxf(mx, row[n]);
        for (int o = 16; o > 0; o >>= 1) mx = fmaxf(mx, __shfl_xor(mx, o, 32));
        float sum = 0.f;
        for (int n = lane; n < N_; n += 32) sum += __expf(row[n] - mx);
        for (int o = 16; o > 0; o >>= 1) sum += __shfl_xor(sum, o, 32);
        float inv = 1.f / sum;
        for (int n = lane; n < N_; n += 32) row[n] = __expf(row[n] - mx) * inv;
      }
    }
    __syncthreads();
    // mha[j] = sum_n attn[h(j)][n] * V[n][j]
    if (tid < D_) {
      const float* ar = sAttn + (tid >> 4) * N_;
      float acc = 0.f;
      for (int n = 0; n < N_; ++n) acc += ar[n] * (float)sV[n * SPAD + tid];
      sMha[tid] = acc;
    }
    __syncthreads();
    // Q2[j] = sum_i Wout[j,i] * mha[i]
    if (tid < D_) {
      const float* wr = wout + tid * 128;
      float acc = 0.f;
      #pragma unroll 4
      for (int i = 0; i < 128; ++i) acc += wr[i] * sMha[i];
      sQ2[tid] = acc;
    }
    __syncthreads();
    // logits[n] = 10*tanh(Q2 . K2[n] / sqrt(128)), masked
    if (tid < N_) {
      const _Float16* kr = sK2 + tid * SPAD;
      float acc = 0.f;
      #pragma unroll 4
      for (int d = 0; d < D_; ++d) acc += (float)kr[d] * sQ2[d];
      float lg = CLIPV * tanhf(acc * 0.08838834764831845f);
      sLogits[tid] = sMask[tid] ? NEGV : lg;
    }
    __syncthreads();
    // block argmax (first-index tie-break, matches jnp.argmax)
    {
      float v = (tid < N_) ? sLogits[tid] : -1e30f;
      int idx = (tid < N_) ? tid : 0x7fffffff;
      sRedF[tid] = v; sRedI[tid] = idx;
      __syncthreads();
      for (int s = 128; s > 0; s >>= 1) {
        if (tid < s) {
          float v2 = sRedF[tid + s]; int i2 = sRedI[tid + s];
          if (v2 > sRedF[tid] || (v2 == sRedF[tid] && i2 < sRedI[tid])) { sRedF[tid] = v2; sRedI[tid] = i2; }
        }
        __syncthreads();
      }
    }
    float mx = sRedF[0];
    int nxtraw = sRedI[0];
    int endv = sScal[0];
    __syncthreads();
    // log-sum-exp
    {
      float p = (tid < N_) ? __expf(sLogits[tid] - mx) : 0.f;
      sRedF[tid] = p; __syncthreads();
      for (int s = 128; s > 0; s >>= 1) { if (tid < s) sRedF[tid] += sRedF[tid + s]; __syncthreads(); }
    }
    float lse = mx + __logf(sRedF[0]);
    int nxt = endv ? nxtraw : 0;
    __syncthreads();
    if (tid == 0) {
      sPi[t] = nxt;
      sChosen[t] = sLogits[nxt] - lse;
      sScal[0] = endv && (nxt != 0);
      sMask[nxt] = 1;
    }
    if (tid < D_) sCtx[2 + tid] = ne[(size_t)(b * N_ + nxt) * D_ + tid];
    __syncthreads();
  }

  // index = first t with pi[t]==0 (else 0)
  {
    int loc = (tid < N_ && sPi[tid] == 0) ? tid : 0x7fffffff;
    sRedI[tid] = loc; __syncthreads();
    for (int s = 128; s > 0; s >>= 1) { if (tid < s) sRedI[tid] = min(sRedI[tid], sRedI[tid + s]); __syncthreads(); }
  }
  int index = sRedI[0];
  if (index == 0x7fffffff) index = 0;
  __syncthreads();
  for (int n = tid; n < N_; n += 256) sSol[n] = 0;
  __syncthreads();
  if (tid == 0) sSol[0] = 1;                      // cols = pi*mask includes 0 entries
  if (tid < N_ && tid < index) sSol[sPi[tid]] = 1;
  __syncthreads();

  // f1/f2 = sol^T x sol
  float f1p = 0.f, f2p = 0.f;
  if (tid < N_ && sSol[tid]) {
    for (int n = 0; n < N_; ++n) {
      if (!sSol[n]) continue;
      f1p += x0b[n * 200 + tid];
      f2p += x1b[n * 200 + tid];
    }
  }
  sRedF[tid] = f1p; __syncthreads();
  for (int s = 128; s > 0; s >>= 1) { if (tid < s) sRedF[tid] += sRedF[tid + s]; __syncthreads(); }
  float f1 = sRedF[0]; __syncthreads();
  sRedF[tid] = f2p; __syncthreads();
  for (int s = 128; s > 0; s >>= 1) { if (tid < s) sRedF[tid] += sRedF[tid + s]; __syncthreads(); }
  float f2 = sRedF[0]; __syncthreads();
  float llp = (tid < N_) ? sChosen[tid] * (sSol[sPi[tid]] ? 1.f : 0.f) : 0.f;
  sRedF[tid] = llp; __syncthreads();
  for (int s = 128; s > 0; s >>= 1) { if (tid < s) sRedF[tid] += sRedF[tid + s]; __syncthreads(); }
  float ll = sRedF[0];

  if (tid == 0) {
    float r0 = rf[0], r1 = rf[1];   // reference uses rf[0,:] for ALL batch elements
    out[b]          = -(f1 * r0 + f2 * r1);
    out[B_ + b]     = f1;
    out[2 * B_ + b] = f2;
    out[3 * B_ + b] = ll;
  }
  for (int t2 = tid; t2 < N_; t2 += 256) out[4 * B_ + b * N_ + t2] = (float)sPi[t2];
}

// ---------------- host ----------------

extern "C" void kernel_launch(void* const* d_in, const int* in_sizes, int n_in,
                              void* d_out, int out_size, void* d_ws, size_t ws_size,
                              hipStream_t stream) {
  (void)in_sizes; (void)n_in; (void)out_size; (void)ws_size;
  const float* x     = (const float*)d_in[0];
  const float* rf    = (const float*)d_in[1];
  const float* ne    = (const float*)d_in[2];
  /* graph_embedding d_in[3] unused by reference */
  const float* wk1   = (const float*)d_in[4];
  const float* wk1b  = (const float*)d_in[5];
  const float* wv    = (const float*)d_in[6];
  const float* wk2   = (const float*)d_in[7];
  const float* wout  = (const float*)d_in[8];
  const float* wq    = (const float*)d_in[9];
  const float* initn = (const float*)d_in[10];
  float* out = (float*)d_out;

  char* ws = (char*)d_ws;
  _Float16* neh = (_Float16*)ws;                         // 512*200*128*2 = 26,214,400 B
  _Float16* wt  = (_Float16*)(ws + 26214400);            // 3*128*128*2  =     98,304 B
  _Float16* kh  = (_Float16*)(ws + 26214400 + 98304);    // 3*26,214,400 = 78,643,200 B

  k_cvt_ne<<<4096, 256, 0, stream>>>(ne, neh, B_ * N_ * D_);
  k_transpose_w<<<192, 256, 0, stream>>>(wk1, wv, wk2, wt);

  dim3 g(1600, 1, 3);
  k_gemm_static<<<g, 128, 0, stream>>>(neh, wt, wk1b, kh);

  const int smem = 177792;  // ~174KB dynamic LDS (< 320KB/WGP)
  (void)hipFuncSetAttribute((const void*)k_decode,
                            hipFuncAttributeMaxDynamicSharedMemorySize, smem);
  k_decode<<<B_, 256, smem, stream>>>(kh, ne, rf, wq, wout, initn, x, out);
}
